// GATv2_43843026157852
// MI455X (gfx1250) — compile-verified, matrix operand order checked by reading, not run on hardware
//
#include <hip/hip_runtime.h>
#include <hip/hip_bf16.h>

// ---------------------------------------------------------------------------
// GATv2 (3 layers) on gfx1250: f16 WMMA GEMMs + atomic segment-softmax passes
// ---------------------------------------------------------------------------

typedef __attribute__((ext_vector_type(16))) _Float16 v16h;
typedef __attribute__((ext_vector_type(8)))  _Float16 v8h;
typedef __attribute__((ext_vector_type(8)))  float    v8f;

#define NN   50000
#define EE   800000
#define ETOT (EE + NN)
#define D0   128
#define OC   40
#define OCP  48
#define NEGS 0.2f

__device__ __forceinline__ float leaky(float v) { return v > 0.f ? v : NEGS * v; }

// float atomic max via signed/unsigned integer ordering trick
__device__ __forceinline__ void atomicMaxF(float* addr, float v) {
    if (v >= 0.f) atomicMax((int*)addr, __float_as_int(v));
    else          atomicMin((unsigned int*)addr, __float_as_uint(v));
}

union V16U { v16h v; v8h h[2]; };

// CDNA5 16-bit A/B fragment: lane l (wave32) holds, for its row/col,
// halves[0..7] = K = kb..kb+7 and halves[8..15] = K = kb+16..kb+23,
// where kb = (l>>4)*8.  Both groups are contiguous 16B in memory when the
// matrix is laid out K-contiguous per row/col -> two b128 loads.
__device__ __forceinline__ v16h ld_frag(const _Float16* p) {
    V16U u;
    u.h[0] = *(const v8h*)(p);
    u.h[1] = *(const v8h*)(p + 16);
    return u.v;
}

// ---------------- f32 -> f16 activation convert ----------------
__global__ void k_cvt_f16(const float* __restrict__ src, _Float16* __restrict__ dst, int n) {
    int t = blockIdx.x * blockDim.x + threadIdx.x;
    if (t < n) dst[t] = (_Float16)src[t];
}

// ---------------- weight prep: transpose [K=128,Nout] -> f16 [NoutPad,128] ----
__global__ void k_prep_w(const float* __restrict__ W, _Float16* __restrict__ Wt,
                         int nout, int noutPad) {
    int t = blockIdx.x * blockDim.x + threadIdx.x;
    if (t >= noutPad * 128) return;
    int n = t >> 7;         // output channel (col)
    int k = t & 127;        // input channel
    float v = (n < nout) ? W[k * nout + n] : 0.f;
    Wt[n * 128 + k] = (_Float16)v;
}

// ---------------- dual GEMM: Cl = A*Wl + bl, Cr = A*Wr + br -------------------
// A: [M,128] f16 row-major.  Wl/Wr: [NoutPad,128] f16 (pre-transposed, K contig).
// One wave per 16x16 output tile; shared A fragment feeds two WMMAs per K-chunk.
__global__ void k_gemm_dual(const _Float16* __restrict__ A,
                            const _Float16* __restrict__ Wl,
                            const _Float16* __restrict__ Wr,
                            const float* __restrict__ bl,
                            const float* __restrict__ br,
                            float* __restrict__ Cl, float* __restrict__ Cr,
                            int ldo, int nvalid) {
    const int l       = threadIdx.x;           // 0..31
    const int rowTile = blockIdx.x;
    const int colTile = blockIdx.y;
    const int r16     = l & 15;
    const int kb      = (l >> 4) * 8;

    const _Float16* arow  = A  + (rowTile * 16 + r16) * 128 + kb;
    const _Float16* wlrow = Wl + (colTile * 16 + r16) * 128 + kb;
    const _Float16* wrrow = Wr + (colTile * 16 + r16) * 128 + kb;

    v8f cl = {};
    v8f cr = {};
#pragma unroll
    for (int kc = 0; kc < 128; kc += 32) {
        v16h a  = ld_frag(arow  + kc);
        v16h bL = ld_frag(wlrow + kc);
        v16h bR = ld_frag(wrrow + kc);
        cl = __builtin_amdgcn_wmma_f32_16x16x32_f16(false, a, false, bL, (short)0, cl, false, false);
        cr = __builtin_amdgcn_wmma_f32_16x16x32_f16(false, a, false, bR, (short)0, cr, false, false);
    }

    // C/D layout: lane l -> col n = l&15; VGPR r -> row (l>>4)*8 + r
    const int col   = colTile * 16 + r16;
    const int mbase = rowTile * 16 + (l >> 4) * 8;
    const float biasL = (col < nvalid) ? bl[col] : 0.f;
    const float biasR = (col < nvalid) ? br[col] : 0.f;
#pragma unroll
    for (int r = 0; r < 8; ++r) {
        Cl[(mbase + r) * ldo + col] = cl[r] + biasL;
        Cr[(mbase + r) * ldo + col] = cr[r] + biasR;
    }
}

// ---------------- generic fill ----------------
__global__ void k_fill(float* __restrict__ p, float v, int n) {
    int t = blockIdx.x * blockDim.x + threadIdx.x;
    if (t < n) p[t] = v;
}

__device__ __forceinline__ void edge_sd(const int* __restrict__ ei, int e, int& s, int& d) {
    if (e < EE) { s = ei[e]; d = ei[EE + e]; }
    else        { s = e - EE; d = s; }          // appended self loops
}

// ---------------- layers 0/1: H=4, C=32, ld=128 ----------------
__global__ void k_edge_score4(const float* __restrict__ xl, const float* __restrict__ xr,
                              const float* __restrict__ att, const int* __restrict__ ei,
                              float* __restrict__ escr, float* __restrict__ emax) {
    int t = blockIdx.x * blockDim.x + threadIdx.x;
    if (t >= ETOT * 4) return;
    int e = t >> 2, h = t & 3, s, d;
    edge_sd(ei, e, s, d);
    const float4* pl = (const float4*)(xl + s * D0 + h * 32);
    const float4* pr = (const float4*)(xr + d * D0 + h * 32);
    const float4* pa = (const float4*)(att + h * 32);
    float acc = 0.f;
#pragma unroll
    for (int i = 0; i < 8; ++i) {
        float4 a = pl[i], b = pr[i], w = pa[i];
        acc += w.x * leaky(a.x + b.x);
        acc += w.y * leaky(a.y + b.y);
        acc += w.z * leaky(a.z + b.z);
        acc += w.w * leaky(a.w + b.w);
    }
    escr[t] = acc;
    atomicMaxF(emax + d * 4 + h, acc);
}

__global__ void k_edge_exp4(const int* __restrict__ ei, float* __restrict__ escr,
                            const float* __restrict__ emax, float* __restrict__ denom) {
    int t = blockIdx.x * blockDim.x + threadIdx.x;
    if (t >= ETOT * 4) return;
    int e = t >> 2, h = t & 3, s, d;
    edge_sd(ei, e, s, d);
    float ee = expf(escr[t] - emax[d * 4 + h]);
    escr[t] = ee;
    atomicAdd(denom + d * 4 + h, ee);
}

__global__ void k_edge_agg4(const float* __restrict__ xl, const int* __restrict__ ei,
                            const float* __restrict__ escr, const float* __restrict__ denom,
                            float* __restrict__ acc) {
    long long t = (long long)blockIdx.x * blockDim.x + threadIdx.x;
    if (t >= (long long)ETOT * D0) return;
    int e = (int)(t >> 7);
    int c = (int)(t & 127);
    int h = c >> 5;
    int s, d;
    edge_sd(ei, e, s, d);
    float alpha = escr[e * 4 + h] / (denom[d * 4 + h] + 1e-16f);
    atomicAdd(acc + d * D0 + c, alpha * xl[s * D0 + c]);
}

// bias + ELU + convert back to f16 for the next GEMM
__global__ void k_finish_elu(const float* __restrict__ acc, const float* __restrict__ bias,
                             _Float16* __restrict__ hf16) {
    int t = blockIdx.x * blockDim.x + threadIdx.x;
    if (t >= NN * D0) return;
    float v = acc[t] + bias[t & 127];
    v = v > 0.f ? v : (expf(v) - 1.f);
    hf16[t] = (_Float16)v;
}

// ---------------- layer 2: H=1, C=40, ld=48 ----------------
__global__ void k_edge_score1(const float* __restrict__ xl, const float* __restrict__ xr,
                              const float* __restrict__ att, const int* __restrict__ ei,
                              float* __restrict__ escr, float* __restrict__ emax) {
    int e = blockIdx.x * blockDim.x + threadIdx.x;
    if (e >= ETOT) return;
    int s, d;
    edge_sd(ei, e, s, d);
    const float4* pl = (const float4*)(xl + s * OCP);
    const float4* pr = (const float4*)(xr + d * OCP);
    const float4* pa = (const float4*)(att);
    float acc = 0.f;
#pragma unroll
    for (int i = 0; i < 10; ++i) {   // 40 channels
        float4 a = pl[i], b = pr[i], w = pa[i];
        acc += w.x * leaky(a.x + b.x);
        acc += w.y * leaky(a.y + b.y);
        acc += w.z * leaky(a.z + b.z);
        acc += w.w * leaky(a.w + b.w);
    }
    escr[e] = acc;
    atomicMaxF(emax + d, acc);
}

__global__ void k_edge_exp1(const int* __restrict__ ei, float* __restrict__ escr,
                            const float* __restrict__ emax, float* __restrict__ denom) {
    int e = blockIdx.x * blockDim.x + threadIdx.x;
    if (e >= ETOT) return;
    int s, d;
    edge_sd(ei, e, s, d);
    float ee = expf(escr[e] - emax[d]);
    escr[e] = ee;
    atomicAdd(denom + d, ee);
}

__global__ void k_edge_agg1(const float* __restrict__ xl, const int* __restrict__ ei,
                            const float* __restrict__ escr, const float* __restrict__ denom,
                            float* __restrict__ acc) {
    long long t = (long long)blockIdx.x * blockDim.x + threadIdx.x;
    if (t >= (long long)ETOT * 64) return;
    int e = (int)(t >> 6);
    int c = (int)(t & 63);
    if (c >= OC) return;
    int s, d;
    edge_sd(ei, e, s, d);
    float alpha = escr[e] / (denom[d] + 1e-16f);
    atomicAdd(acc + d * OC + c, alpha * xl[s * OCP + c]);
}

__global__ void k_logsoftmax(const float* __restrict__ acc, const float* __restrict__ bias,
                             float* __restrict__ out) {
    int n = blockIdx.x * blockDim.x + threadIdx.x;
    if (n >= NN) return;
    const float* row = acc + n * OC;
    float m = -3.0e38f;
    for (int c = 0; c < OC; ++c) m = fmaxf(m, row[c] + bias[c]);
    float ssum = 0.f;
    for (int c = 0; c < OC; ++c) ssum += expf(row[c] + bias[c] - m);
    float lg = logf(ssum);
    float* orow = out + n * OC;
    for (int c = 0; c < OC; ++c) orow[c] = (row[c] + bias[c]) - m - lg;
}

// ---------------------------------------------------------------------------
extern "C" void kernel_launch(void* const* d_in, const int* in_sizes, int n_in,
                              void* d_out, int out_size, void* d_ws, size_t ws_size,
                              hipStream_t stream) {
    const float* x    = (const float*)d_in[0];
    const int*   ei   = (const int*)d_in[1];
    const float* Wl0  = (const float*)d_in[2];
    const float* bl0  = (const float*)d_in[3];
    const float* Wr0  = (const float*)d_in[4];
    const float* br0  = (const float*)d_in[5];
    const float* att0 = (const float*)d_in[6];
    const float* b0   = (const float*)d_in[7];
    const float* Wl1  = (const float*)d_in[8];
    const float* bl1  = (const float*)d_in[9];
    const float* Wr1  = (const float*)d_in[10];
    const float* br1  = (const float*)d_in[11];
    const float* att1 = (const float*)d_in[12];
    const float* b1   = (const float*)d_in[13];
    const float* Wl2  = (const float*)d_in[14];
    const float* bl2  = (const float*)d_in[15];
    const float* Wr2  = (const float*)d_in[16];
    const float* br2  = (const float*)d_in[17];
    const float* att2 = (const float*)d_in[18];
    const float* b2   = (const float*)d_in[19];
    float* out = (float*)d_out;

    // workspace layout (256B-aligned slices)
    char*  ws = (char*)d_ws;
    size_t o  = 0;
    auto take = [&](size_t bytes) -> void* {
        void* p = ws + o;
        o += (bytes + 255) & ~(size_t)255;
        return p;
    };
    _Float16* hf16  = (_Float16*)take((size_t)NN * D0 * 2);
    _Float16* wt0l  = (_Float16*)take((size_t)D0 * D0 * 2);
    _Float16* wt0r  = (_Float16*)take((size_t)D0 * D0 * 2);
    _Float16* wt1l  = (_Float16*)take((size_t)D0 * D0 * 2);
    _Float16* wt1r  = (_Float16*)take((size_t)D0 * D0 * 2);
    _Float16* wt2l  = (_Float16*)take((size_t)OCP * D0 * 2);
    _Float16* wt2r  = (_Float16*)take((size_t)OCP * D0 * 2);
    float*    xl    = (float*)take((size_t)NN * D0 * 4);
    float*    xr    = (float*)take((size_t)NN * D0 * 4);
    float*    accb  = (float*)take((size_t)NN * D0 * 4);
    float*    escr  = (float*)take((size_t)ETOT * 4 * 4);
    float*    emax  = (float*)take((size_t)NN * 4 * 4);
    float*    denom = (float*)take((size_t)NN * 4 * 4);

    const int B = 256;
    auto blocks = [](long long n, int b) { return (unsigned)((n + b - 1) / b); };

    // ---- prep: x -> f16, weights -> transposed f16 ----
    k_cvt_f16<<<blocks((long long)NN * D0, B), B, 0, stream>>>(x, hf16, NN * D0);
    k_prep_w<<<blocks(D0 * 128, B), B, 0, stream>>>(Wl0, wt0l, D0, D0);
    k_prep_w<<<blocks(D0 * 128, B), B, 0, stream>>>(Wr0, wt0r, D0, D0);
    k_prep_w<<<blocks(D0 * 128, B), B, 0, stream>>>(Wl1, wt1l, D0, D0);
    k_prep_w<<<blocks(D0 * 128, B), B, 0, stream>>>(Wr1, wt1r, D0, D0);
    k_prep_w<<<blocks(OCP * 128, B), B, 0, stream>>>(Wl2, wt2l, OC, OCP);
    k_prep_w<<<blocks(OCP * 128, B), B, 0, stream>>>(Wr2, wt2r, OC, OCP);

    const _Float16* wtl[2] = { wt0l, wt1l };
    const _Float16* wtr[2] = { wt0r, wt1r };
    const float* blv[2] = { bl0, bl1 };
    const float* brv[2] = { br0, br1 };
    const float* attv[2] = { att0, att1 };
    const float* bv[2] = { b0, b1 };

    // ---- layers 0 and 1 (H=4, C=32, concat) ----
    for (int L = 0; L < 2; ++L) {
        dim3 g(NN / 16, D0 / 16);
        k_gemm_dual<<<g, 32, 0, stream>>>(hf16, wtl[L], wtr[L], blv[L], brv[L],
                                          xl, xr, D0, D0);
        k_fill<<<blocks(NN * 4, B), B, 0, stream>>>(emax, -3.0e38f, NN * 4);
        k_fill<<<blocks(NN * 4, B), B, 0, stream>>>(denom, 0.f, NN * 4);
        k_fill<<<blocks((long long)NN * D0, B), B, 0, stream>>>(accb, 0.f, NN * D0);
        k_edge_score4<<<blocks((long long)ETOT * 4, B), B, 0, stream>>>(xl, xr, attv[L], ei, escr, emax);
        k_edge_exp4<<<blocks((long long)ETOT * 4, B), B, 0, stream>>>(ei, escr, emax, denom);
        k_edge_agg4<<<blocks((long long)ETOT * D0, B), B, 0, stream>>>(xl, ei, escr, denom, accb);
        k_finish_elu<<<blocks((long long)NN * D0, B), B, 0, stream>>>(accb, bv[L], hf16);
    }

    // ---- layer 2 (H=1, C=40, mean==identity) ----
    {
        dim3 g(NN / 16, OCP / 16);
        k_gemm_dual<<<g, 32, 0, stream>>>(hf16, wt2l, wt2r, bl2, br2, xl, xr, OCP, OC);
        k_fill<<<blocks(NN, B), B, 0, stream>>>(emax, -3.0e38f, NN);
        k_fill<<<blocks(NN, B), B, 0, stream>>>(denom, 0.f, NN);
        k_fill<<<blocks((long long)NN * OC, B), B, 0, stream>>>(accb, 0.f, NN * OC);
        k_edge_score1<<<blocks(ETOT, B), B, 0, stream>>>(xl, xr, att2, ei, escr, emax);
        k_edge_exp1<<<blocks(ETOT, B), B, 0, stream>>>(ei, escr, emax, denom);
        k_edge_agg1<<<blocks((long long)ETOT * 64, B), B, 0, stream>>>(xl, ei, escr, denom, accb);
        k_logsoftmax<<<blocks(NN, B), B, 0, stream>>>(accb, b2, out);
    }
}
